// DeepFCNet_21560735826191
// MI455X (gfx1250) — matrix operand order repaired and checked by reading, not compiled, not run on hardware
//
#include <hip/hip_runtime.h>
#include <hip/hip_bf16.h>
#include <math.h>

// ---------------------------------------------------------------------------
// Problem constants (from the reference): B=32, P=9045, F=750
//   rows = B*P = 289440 = 18090 tiles of 16
//   Layer-1 K padded 750 -> 768 = 24 chunks of 32
//   Classifier K padded 9045 -> 9056 = 283 chunks of 32
// ---------------------------------------------------------------------------
#define N_ROWS   289440
#define N_TILES  18090
#define KC1      24          // layer-1 K chunks
#define F_DIM    750
#define P_DIM    9045
#define KC_CLS   283         // classifier K chunks

typedef __attribute__((ext_vector_type(16))) __bf16 v16bf;
typedef __attribute__((ext_vector_type(8)))  float  v8f;

__device__ __forceinline__ v8f wmma_bf16(v16bf a, v16bf b, v8f c) {
  return __builtin_amdgcn_wmma_f32_16x16x32_bf16(false, a, false, b,
                                                 (short)0, c, false, false);
}

// ---------------------------------------------------------------------------
// Prep: repack similarity weights into bf16 WMMA-B fragment order.
// B-matrix (32x16, 16-bit) per-lane layout: lane holds column n = lane&15,
// K values = (j>>3)*16 + (lane>>4)*8 + (j&7)  (j = 0..15)   [ISA 7.12.x B layout]
//  w1f: [kc][nt][lane][16]  (24 chunks x 2 n-tiles, K padded to 768)
//  w2f: [lane][16]          (32x16 exactly)
//  w3f: [lane][16]          (16x8 padded to 32x16 with zeros)
// ---------------------------------------------------------------------------
__global__ void prep_frags(const float* __restrict__ sw1,
                           const float* __restrict__ sw2,
                           const float* __restrict__ sw3,
                           __bf16* __restrict__ w1f,
                           __bf16* __restrict__ w2f,
                           __bf16* __restrict__ w3f) {
  int tid    = blockIdx.x * blockDim.x + threadIdx.x;
  int stride = blockDim.x * gridDim.x;
  // w1f: 24*2*32 = 1536 lane-entries
  for (int e = tid; e < KC1 * 2 * 32; e += stride) {
    int lane = e & 31;
    int nt   = (e >> 5) & 1;
    int kc   = e >> 6;
    int half = lane >> 4;
    int n    = nt * 16 + (lane & 15);
    for (int j = 0; j < 16; ++j) {
      int k = kc * 32 + (j >> 3) * 16 + half * 8 + (j & 7);
      w1f[e * 16 + j] = (k < F_DIM) ? (__bf16)sw1[k * 32 + n] : (__bf16)0.0f;
    }
  }
  // w2f: 32 lane-entries (K=32, N=16 exact)
  for (int e = tid; e < 32; e += stride) {
    int half = e >> 4, n = e & 15;
    for (int j = 0; j < 16; ++j) {
      int k = (j >> 3) * 16 + half * 8 + (j & 7);
      w2f[e * 16 + j] = (__bf16)sw2[k * 16 + n];
    }
  }
  // w3f: 32 lane-entries, sw3 is 16x8 -> zero-pad K 16..31, N 8..15
  for (int e = tid; e < 32; e += stride) {
    int half = e >> 4, n = e & 15;
    for (int j = 0; j < 16; ++j) {
      int k = (j >> 3) * 16 + half * 8 + (j & 7);
      float v = (k < 16 && n < 8) ? sw3[k * 8 + n] : 0.0f;
      w3f[e * 16 + j] = (__bf16)v;
    }
  }
}

// ---------------------------------------------------------------------------
// Similarity MLP: one wave per 16-row tile.
//   L1: 16x750 @ 750x32 via 24x2 WMMA bf16 (f32 accumulate) + bias + relu
//   L2: transpose via per-wave LDS, 1 WMMA (K=32) + bias + relu
//   L3: transpose via per-wave LDS (K padded: zeroed cols), 1 WMMA
//   L4: scale by sw4, shfl_xor butterfly reduce, tanh, store sim row
// A-matrix (16x32, 16-bit) per-lane layout: lane holds row m = lane&15,
// K(j) = (j>>3)*16 + (lane>>4)*8 + (j&7)
// ---------------------------------------------------------------------------
__global__ void __launch_bounds__(256)
sim_kernel(const float* __restrict__ x,
           const __bf16* __restrict__ w1f,
           const __bf16* __restrict__ w2f,
           const __bf16* __restrict__ w3f,
           const float* __restrict__ sb1, const float* __restrict__ sb2,
           const float* __restrict__ sb3, const float* __restrict__ sw4,
           const float* __restrict__ sb4,
           float* __restrict__ simBuf) {
  __shared__ __bf16 h1s[8][16][32];   // per-wave 1KB region
  __shared__ __bf16 h2s[8][16][32];   // per-wave 1KB region (cols 16..31 == 0)

  const int warp = threadIdx.x >> 5;
  const int lane = threadIdx.x & 31;
  const int half = lane >> 4;
  const int n0   = lane & 15;

  const int tile = blockIdx.x * 8 + warp;
  if (tile >= N_TILES) return;        // tile is wave-uniform: whole wave exits

  // zero the K-padding columns of the h2 staging region (once per wave)
#pragma unroll
  for (int r = 0; r < 8; ++r)
    h2s[warp][r + half * 8][16 + n0] = (__bf16)0.0f;

  // layer-2/3 B fragments (resident for the whole tile)
  v16bf b2 = *(const v16bf*)(w2f + lane * 16);
  v16bf b3 = *(const v16bf*)(w3f + lane * 16);

  const float bias1a = sb1[n0];
  const float bias1b = sb1[16 + n0];
  const float bias2  = sb2[n0];
  const float bias3  = (n0 < 8) ? sb3[n0] : 0.0f;
  const float w4v    = (n0 < 8) ? sw4[n0] : 0.0f;
  const float b4v    = sb4[0];

  const float* xrow = x + (size_t)(tile * 16 + n0) * F_DIM;  // A row m = lane&15

  // ---- layer 1: 24 chunks of K=32, two N-tiles -------------------------
  v8f acc0 = {}; v8f acc1 = {};
  for (int kc = 0; kc < KC1; ++kc) {
    __builtin_prefetch((const void*)(xrow + kc * 32 + 64), 0, 0);
    v16bf a;
#pragma unroll
    for (int g = 0; g < 2; ++g) {
      const int kb = kc * 32 + g * 16 + half * 8;
#pragma unroll
      for (int u = 0; u < 4; ++u) {
        const int k = kb + 2 * u;                 // even; F_DIM even too
        float2 v;
        if (k < F_DIM) v = *(const float2*)(xrow + k);
        else { v.x = 0.0f; v.y = 0.0f; }
        a[g * 8 + 2 * u]     = (__bf16)v.x;
        a[g * 8 + 2 * u + 1] = (__bf16)v.y;
      }
    }
    const __bf16* bp = w1f + ((size_t)(kc * 2) * 32 + lane) * 16;
    v16bf b0 = *(const v16bf*)bp;
    v16bf b1 = *(const v16bf*)(bp + 512);         // nt=1 entry is +32*16 elems
    acc0 = wmma_bf16(a, b0, acc0);
    acc1 = wmma_bf16(a, b1, acc1);
  }

  // bias + relu -> h1 (16x32 bf16) into LDS (C layout: lane holds column n0)
#pragma unroll
  for (int r = 0; r < 8; ++r) {
    const int m = r + half * 8;
    h1s[warp][m][n0]      = (__bf16)fmaxf(acc0[r] + bias1a, 0.0f);
    h1s[warp][m][16 + n0] = (__bf16)fmaxf(acc1[r] + bias1b, 0.0f);
  }
  asm volatile("s_wait_dscnt 0x0" ::: "memory");  // wave-local LDS transpose sync

  // ---- layer 2: A from LDS (row m = n0), one WMMA ----------------------
  v16bf a2;
#pragma unroll
  for (int j = 0; j < 16; ++j) {
    const int kl = (j >> 3) * 16 + half * 8 + (j & 7);
    a2[j] = h1s[warp][n0][kl];
  }
  v8f acc2 = {};
  acc2 = wmma_bf16(a2, b2, acc2);

#pragma unroll
  for (int r = 0; r < 8; ++r) {
    const int m = r + half * 8;
    h2s[warp][m][n0] = (__bf16)fmaxf(acc2[r] + bias2, 0.0f);
  }
  asm volatile("s_wait_dscnt 0x0" ::: "memory");

  // ---- layer 3: A from LDS (K padded with zero cols), one WMMA ---------
  v16bf a3;
#pragma unroll
  for (int j = 0; j < 16; ++j) {
    const int kl = (j >> 3) * 16 + half * 8 + (j & 7);
    a3[j] = h2s[warp][n0][kl];
  }
  v8f acc3 = {};
  acc3 = wmma_bf16(a3, b3, acc3);

  // ---- layer 4: h3 * sw4, reduce across the 16 columns, tanh -----------
  float s[8];
#pragma unroll
  for (int r = 0; r < 8; ++r) {
    float h3 = (n0 < 8) ? fmaxf(acc3[r] + bias3, 0.0f) : 0.0f;
    float p  = h3 * w4v;
    p += __shfl_xor(p, 1, 16);
    p += __shfl_xor(p, 2, 16);
    p += __shfl_xor(p, 4, 16);
    p += __shfl_xor(p, 8, 16);
    s[r] = p;
  }
  if (n0 == 0) {
#pragma unroll
    for (int r = 0; r < 8; ++r)
      simBuf[tile * 16 + half * 8 + r] = tanhf(s[r] + b4v);
  }
}

// ---------------------------------------------------------------------------
// Classifier layer 1: c1 = relu(sim[32x9045] @ cw1[9045x1024] + cb1)
// 2 M-tiles x 64 N-tiles; one wave per tile; bf16 conversion on the fly.
// ---------------------------------------------------------------------------
__global__ void __launch_bounds__(256)
cls1_kernel(const float* __restrict__ simBuf,
            const float* __restrict__ cw1,
            const float* __restrict__ cb1,
            float* __restrict__ c1buf) {
  const int warp = threadIdx.x >> 5;
  const int lane = threadIdx.x & 31;
  const int tid  = blockIdx.x * 8 + warp;       // 0..127
  if (tid >= 128) return;
  const int mt = tid >> 6, nt = tid & 63;
  const int half = lane >> 4, n0 = lane & 15;
  const int subj = mt * 16 + n0;                // A row (subject)
  const int n    = nt * 16 + n0;                // B column
  const float* srow = simBuf + (size_t)subj * P_DIM;

  v8f acc = {};
  for (int kc = 0; kc < KC_CLS; ++kc) {
    v16bf a, b;
#pragma unroll
    for (int j = 0; j < 16; ++j) {
      const int ka = kc * 32 + (j >> 3) * 16 + half * 8 + (j & 7); // A layout
      const int kb = kc * 32 + (j >> 3) * 16 + half * 8 + (j & 7); // B layout
      a[j] = (__bf16)((ka < P_DIM) ? srow[ka] : 0.0f);
      b[j] = (__bf16)((kb < P_DIM) ? cw1[(size_t)kb * 1024 + n] : 0.0f);
    }
    acc = wmma_bf16(a, b, acc);
  }
  const float bias = cb1[n];
#pragma unroll
  for (int r = 0; r < 8; ++r) {
    const int m = mt * 16 + r + half * 8;       // subject
    c1buf[m * 1024 + n] = fmaxf(acc[r] + bias, 0.0f);
  }
}

// ---------------------------------------------------------------------------
// Classifier tail: 1024 -> 256 -> 64 -> 3 + log_softmax. One block/subject.
// Weight reads are coalesced across threads (row-major [K][N] layouts).
// ---------------------------------------------------------------------------
__global__ void __launch_bounds__(256)
cls_tail_kernel(const float* __restrict__ c1buf,
                const float* __restrict__ cw2, const float* __restrict__ cb2,
                const float* __restrict__ cw3, const float* __restrict__ cb3,
                const float* __restrict__ cw4, const float* __restrict__ cb4,
                float* __restrict__ out) {
  __shared__ float l1[1024];
  __shared__ float l2[256];
  __shared__ float l3[64];
  __shared__ float lg[3];
  const int b = blockIdx.x;
  const int t = threadIdx.x;

  for (int i = t; i < 1024; i += 256) l1[i] = c1buf[b * 1024 + i];
  __syncthreads();

  {
    float acc = cb2[t];
    for (int k = 0; k < 1024; ++k) acc = fmaf(l1[k], cw2[k * 256 + t], acc);
    l2[t] = fmaxf(acc, 0.0f);
  }
  __syncthreads();

  if (t < 64) {
    float acc = cb3[t];
    for (int k = 0; k < 256; ++k) acc = fmaf(l2[k], cw3[k * 64 + t], acc);
    l3[t] = fmaxf(acc, 0.0f);
  }
  __syncthreads();

  if (t < 3) {
    float acc = cb4[t];
    for (int k = 0; k < 64; ++k) acc = fmaf(l3[k], cw4[k * 3 + t], acc);
    lg[t] = acc;
  }
  __syncthreads();

  if (t < 3) {
    float m = fmaxf(lg[0], fmaxf(lg[1], lg[2]));
    float s = expf(lg[0] - m) + expf(lg[1] - m) + expf(lg[2] - m);
    out[b * 3 + t] = (lg[t] - m) - logf(s);
  }
}

// ---------------------------------------------------------------------------
// Launch
// ---------------------------------------------------------------------------
extern "C" void kernel_launch(void* const* d_in, const int* in_sizes, int n_in,
                              void* d_out, int out_size, void* d_ws, size_t ws_size,
                              hipStream_t stream) {
  const float* x   = (const float*)d_in[0];
  const float* sw1 = (const float*)d_in[1];
  const float* sb1 = (const float*)d_in[2];
  const float* sw2 = (const float*)d_in[3];
  const float* sb2 = (const float*)d_in[4];
  const float* sw3 = (const float*)d_in[5];
  const float* sb3 = (const float*)d_in[6];
  const float* sw4 = (const float*)d_in[7];
  const float* sb4 = (const float*)d_in[8];
  const float* cw1 = (const float*)d_in[9];
  const float* cb1 = (const float*)d_in[10];
  const float* cw2 = (const float*)d_in[11];
  const float* cb2 = (const float*)d_in[12];
  const float* cw3 = (const float*)d_in[13];
  const float* cb3 = (const float*)d_in[14];
  const float* cw4 = (const float*)d_in[15];
  const float* cb4 = (const float*)d_in[16];

  // Workspace layout (bytes), 256-aligned sections. Total ~1.28 MB.
  char* ws = (char*)d_ws;
  __bf16* w1f    = (__bf16*)(ws + 0);        //  49152 B
  __bf16* w2f    = (__bf16*)(ws + 49152);    //   1024 B
  __bf16* w3f    = (__bf16*)(ws + 50176);    //   1024 B
  float*  simBuf = (float*)(ws + 51200);     // 1157760 B (289440 f32)
  float*  c1buf  = (float*)(ws + 1209088);   // 131072 B (32*1024 f32)

  prep_frags<<<4, 256, 0, stream>>>(sw1, sw2, sw3, w1f, w2f, w3f);

  const int blocksA = (N_TILES + 7) / 8;     // one 16-row tile per wave
  sim_kernel<<<blocksA, 256, 0, stream>>>(x, w1f, w2f, w3f,
                                          sb1, sb2, sb3, sw4, sb4, simBuf);

  cls1_kernel<<<16, 256, 0, stream>>>(simBuf, cw1, cb1, c1buf);

  cls_tail_kernel<<<32, 256, 0, stream>>>(c1buf, cw2, cb2, cw3, cb3,
                                          cw4, cb4, (float*)d_out);
}
